// MPNNLayer_30382598652103
// MI455X (gfx1250) — matrix operand order repaired and checked by reading, not compiled, not run on hardware
//
#include <hip/hip_runtime.h>
#include <hip/hip_bf16.h>

typedef __bf16 bf16;
typedef __attribute__((ext_vector_type(8)))  float  v8f;
typedef __attribute__((ext_vector_type(16))) __bf16 v16bf;

#define NN 50000
#define EE 800000
#define CC 64

// ---------------- fragment builders (wave32 WMMA layouts, ISA 7.12.2) -------

// A fragment (16x32 bf16, MxK) from row-major bf16 row pointer.
// lane L: M = L&15 ; lanes 0-15 hold K = {0..7,16..23}, lanes 16-31 K = {8..15,24..31}
__device__ __forceinline__ v16bf a_frag_bf16(const bf16* rowp, int koff, int lane) {
    int kb = (lane >> 4) << 3;              // 0 or 8
    const bf16* p = rowp + koff + kb;
    v16bf a;
#pragma unroll
    for (int i = 0; i < 8; ++i) { a[i] = p[i]; a[i + 8] = p[i + 16]; }
    return a;
}

// Same, but source is f32 row-major (convert on the fly).
__device__ __forceinline__ v16bf a_frag_f32(const float* rowp, int koff, int lane) {
    int kb = (lane >> 4) << 3;
    const float* p = rowp + koff + kb;
    v16bf a;
#pragma unroll
    for (int i = 0; i < 8; ++i) { a[i] = (bf16)p[i]; a[i + 8] = (bf16)p[i + 16]; }
    return a;
}

// B fragment (32x16 bf16, KxN) from COLUMN-major weights Wt[n][k] (stride Ktot).
// lane L: N = L&15 ; lanes 0-15 hold K = {0..7,16..23}, lanes 16-31 K = {8..15,24..31}
__device__ __forceinline__ v16bf b_frag(const bf16* wt, int Ktot, int n, int koff, int lane) {
    const bf16* p = wt + (long)n * Ktot + koff + ((lane >> 4) << 3);
    v16bf b;
#pragma unroll
    for (int i = 0; i < 8; ++i) { b[i] = p[i]; b[i + 8] = p[i + 16]; }
    return b;
}

#define WMMA_BF16(a, b, c) \
    __builtin_amdgcn_wmma_f32_16x16x32_bf16(false, (a), false, (b), (short)0, (c), false, false)

// ---------------- prep: bf16 weight transposes, h2 concat, m_sum zero -------

__global__ __launch_bounds__(256) void prep_kernel(
    const float* __restrict__ h, const float* __restrict__ rnf,
    const float* __restrict__ W1, const float* __restrict__ W2,
    const float* __restrict__ U1, const float* __restrict__ U2,
    bf16* __restrict__ h2bf, float* __restrict__ m_sum,
    bf16* __restrict__ W1t, bf16* __restrict__ W2t,
    bf16* __restrict__ U1t, bf16* __restrict__ U2t) {
    long tid = (long)blockIdx.x * blockDim.x + threadIdx.x;
    long stride = (long)gridDim.x * blockDim.x;

    // h2 = concat(h, rnf) -> bf16 [N,128]
    for (long i = tid; i < (long)NN * 128; i += stride) {
        long node = i >> 7; int c = (int)(i & 127);
        float v = (c < CC) ? h[node * CC + c] : rnf[node * CC + (c - CC)];
        h2bf[i] = (bf16)v;
    }
    // zero accumulation buffer
    for (long i = tid; i < (long)NN * CC; i += stride) m_sum[i] = 0.0f;
    // W1t[n][k] = W1[k][n]  (W1 row-major [320][64])
    for (long i = tid; i < 320L * 64; i += stride) {
        int n = (int)(i / 320), k = (int)(i % 320);
        W1t[i] = (bf16)W1[(long)k * 64 + n];
    }
    for (long i = tid; i < 64L * 64; i += stride) {
        int n = (int)(i / 64), k = (int)(i % 64);
        W2t[i] = (bf16)W2[(long)k * 64 + n];
    }
    for (long i = tid; i < 128L * 64; i += stride) {
        int n = (int)(i / 128), k = (int)(i % 128);
        U1t[i] = (bf16)U1[(long)k * 64 + n];
    }
    for (long i = tid; i < 64L * 64; i += stride) {
        int n = (int)(i / 64), k = (int)(i % 64);
        U2t[i] = (bf16)U2[(long)k * 64 + n];
    }
}

// ---------------- edge kernel: m = relu(msg_in@W1+b1)@W2+b2 ; scatter-add ---

__global__ __launch_bounds__(256) void edge_kernel(
    const bf16* __restrict__ h2bf, const float* __restrict__ e,
    const int* __restrict__ src, const int* __restrict__ dst,
    const bf16* __restrict__ W1t, const float* __restrict__ b1,
    const bf16* __restrict__ W2t, const float* __restrict__ b2,
    float* __restrict__ m_sum) {
    __shared__ bf16 lds[8 * 16 * 64];           // per-wave 16x64 bf16 staging
    int wave = threadIdx.x >> 5;
    int lane = threadIdx.x & 31;
    long tile = (long)blockIdx.x * 8 + wave;    // 50000 tiles exactly
    long ebase = tile * 16;

    int row = lane & 15;
    int hi = (lane >> 4) << 3;                  // 0 or 8 (C/D row group)
    int ncol0 = lane & 15;

    int s = src[ebase + row];
    int d = dst[ebase + row];
    const bf16*  hs = h2bf + (long)s * 128;
    const bf16*  hd = h2bf + (long)d * 128;
    const float* er = e + (ebase + row) * (long)CC;

    // GEMM1: [16,320] x [320,64], K = 10 tiles of 32
    v8f acc[4] = {v8f{0}, v8f{0}, v8f{0}, v8f{0}};
#pragma unroll
    for (int kt = 0; kt < 10; ++kt) {
        v16bf a;
        if (kt < 4)      a = a_frag_bf16(hs, kt * 32, lane);          // src h2 (K 0..127)
        else if (kt < 6) a = a_frag_f32(er, (kt - 4) * 32, lane);     // e     (K 128..191)
        else             a = a_frag_bf16(hd, (kt - 6) * 32, lane);    // dst h2 (K 192..319)
#pragma unroll
        for (int nt = 0; nt < 4; ++nt)
            acc[nt] = WMMA_BF16(a, b_frag(W1t, 320, nt * 16 + ncol0, kt * 32, lane), acc[nt]);
    }

    // bias + relu, re-shape via LDS to A-layout (same-wave DS ops are in-order)
    bf16* L = lds + wave * (16 * 64);
#pragma unroll
    for (int nt = 0; nt < 4; ++nt) {
        int col = nt * 16 + ncol0;
        float bias = b1[col];
#pragma unroll
        for (int j = 0; j < 8; ++j) {
            float v = acc[nt][j] + bias;
            L[(j + hi) * 64 + col] = (bf16)(v > 0.0f ? v : 0.0f);
        }
    }
    asm volatile("s_wait_dscnt 0" ::: "memory");

    // GEMM2: [16,64] x [64,64], K = 2 tiles
    v8f out[4] = {v8f{0}, v8f{0}, v8f{0}, v8f{0}};
#pragma unroll
    for (int kt = 0; kt < 2; ++kt) {
        v16bf a2 = a_frag_bf16(L + row * 64, kt * 32, lane);
#pragma unroll
        for (int nt = 0; nt < 4; ++nt)
            out[nt] = WMMA_BF16(a2, b_frag(W2t, 64, nt * 16 + ncol0, kt * 32, lane), out[nt]);
    }

    // scatter-add messages into m_sum[dst]
#pragma unroll
    for (int nt = 0; nt < 4; ++nt) {
        int col = nt * 16 + ncol0;
        float bias = b2[col];
#pragma unroll
        for (int j = 0; j < 8; ++j) {
            int m  = j + hi;
            int dn = dst[ebase + m];
            atomicAdd(&m_sum[(long)dn * CC + col], out[nt][j] + bias);
        }
    }
}

// ---------------- node kernel: h_new = relu([m_sum|h]@U1+c1)@U2+c2 ----------

__global__ __launch_bounds__(256) void node_kernel(
    const float* __restrict__ m_sum, const float* __restrict__ h,
    const bf16* __restrict__ U1t, const float* __restrict__ c1,
    const bf16* __restrict__ U2t, const float* __restrict__ c2,
    float* __restrict__ hnew) {
    __shared__ bf16 lds[8 * 16 * 64];
    int wave = threadIdx.x >> 5;
    int lane = threadIdx.x & 31;
    long tile = (long)blockIdx.x * 8 + wave;
    if (tile >= (NN / 16)) return;              // 3125 full tiles
    long nbase = tile * 16;

    int row = lane & 15;
    int hi = (lane >> 4) << 3;
    int ncol0 = lane & 15;
    long node = nbase + row;

    v8f acc[4] = {v8f{0}, v8f{0}, v8f{0}, v8f{0}};
#pragma unroll
    for (int kt = 0; kt < 4; ++kt) {
        v16bf a;
        if (kt < 2) a = a_frag_f32(m_sum + node * CC, kt * 32, lane);      // K 0..63
        else        a = a_frag_f32(h + node * CC, (kt - 2) * 32, lane);    // K 64..127
#pragma unroll
        for (int nt = 0; nt < 4; ++nt)
            acc[nt] = WMMA_BF16(a, b_frag(U1t, 128, nt * 16 + ncol0, kt * 32, lane), acc[nt]);
    }

    bf16* L = lds + wave * (16 * 64);
#pragma unroll
    for (int nt = 0; nt < 4; ++nt) {
        int col = nt * 16 + ncol0;
        float bias = c1[col];
#pragma unroll
        for (int j = 0; j < 8; ++j) {
            float v = acc[nt][j] + bias;
            L[(j + hi) * 64 + col] = (bf16)(v > 0.0f ? v : 0.0f);
        }
    }
    asm volatile("s_wait_dscnt 0" ::: "memory");

    v8f out[4] = {v8f{0}, v8f{0}, v8f{0}, v8f{0}};
#pragma unroll
    for (int kt = 0; kt < 2; ++kt) {
        v16bf a2 = a_frag_bf16(L + row * 64, kt * 32, lane);
#pragma unroll
        for (int nt = 0; nt < 4; ++nt)
            out[nt] = WMMA_BF16(a2, b_frag(U2t, 64, nt * 16 + ncol0, kt * 32, lane), out[nt]);
    }

#pragma unroll
    for (int nt = 0; nt < 4; ++nt) {
        int col = nt * 16 + ncol0;
        float bias = c2[col];
#pragma unroll
        for (int j = 0; j < 8; ++j) {
            long nd = nbase + j + hi;
            hnew[nd * CC + col] = out[nt][j] + bias;
        }
    }
}

// ---------------- e passthrough ---------------------------------------------

__global__ __launch_bounds__(256) void copy_e_kernel(const float4* __restrict__ e4,
                                                     float4* __restrict__ out4, long n4) {
    long tid = (long)blockIdx.x * blockDim.x + threadIdx.x;
    long stride = (long)gridDim.x * blockDim.x;
    for (long i = tid; i < n4; i += stride) out4[i] = e4[i];
}

// ---------------- launch -----------------------------------------------------

extern "C" void kernel_launch(void* const* d_in, const int* in_sizes, int n_in,
                              void* d_out, int out_size, void* d_ws, size_t ws_size,
                              hipStream_t stream) {
    const float* h   = (const float*)d_in[0];
    const float* e   = (const float*)d_in[1];
    const float* rnf = (const float*)d_in[2];
    const int*   src = (const int*)d_in[3];
    const int*   dst = (const int*)d_in[4];
    const float* W1  = (const float*)d_in[5];
    const float* b1  = (const float*)d_in[6];
    const float* W2  = (const float*)d_in[7];
    const float* b2  = (const float*)d_in[8];
    const float* U1  = (const float*)d_in[9];
    const float* c1  = (const float*)d_in[10];
    const float* U2  = (const float*)d_in[11];
    const float* c2  = (const float*)d_in[12];

    float* hnew  = (float*)d_out;                 // [N,64]
    float* e_out = (float*)d_out + (long)NN * CC; // [E,64]

    // workspace layout
    char* ws = (char*)d_ws;
    bf16*  h2bf  = (bf16*)(ws);                        // N*128 bf16 = 12.8 MB
    float* m_sum = (float*)(ws + 12800000);            // N*64 f32  = 12.8 MB
    bf16*  W1t   = (bf16*)(ws + 25600000);             // 64x320
    bf16*  W2t   = (bf16*)(ws + 25600000 + 40960);     // 64x64
    bf16*  U1t   = (bf16*)(ws + 25600000 + 40960 + 8192);          // 64x128
    bf16*  U2t   = (bf16*)(ws + 25600000 + 40960 + 8192 + 16384);  // 64x64

    prep_kernel<<<2048, 256, 0, stream>>>(h, rnf, W1, W2, U1, U2,
                                          h2bf, m_sum, W1t, W2t, U1t, U2t);

    // 800000 edges / 16 per wave / 8 waves per block = 6250 blocks (exact)
    edge_kernel<<<6250, 256, 0, stream>>>(h2bf, e, src, dst, W1t, b1, W2t, b2, m_sum);

    // 3125 node tiles / 8 waves per block = 391 blocks (last partially active)
    node_kernel<<<391, 256, 0, stream>>>(m_sum, h, U1t, c1, U2t, c2, hnew);

    copy_e_kernel<<<4096, 256, 0, stream>>>((const float4*)e, (float4*)e_out,
                                            (long)EE * CC / 4);
}
